// DNC_59983513256103
// MI455X (gfx1250) — matrix operand order repaired and checked by reading, not compiled, not run on hardware
//
#include <hip/hip_runtime.h>
#include <hip/hip_bf16.h>

// Problem sizes (from reference)
#define NN    8192
#define WD    64
#define RRR   4
#define OUTD  256
#define IND   256
#define CCn   727      // C = OUT + IFACE
#define G4    2908     // 4*C

typedef _Float16 half_t;
typedef _Float16 v16h __attribute__((ext_vector_type(16)));
typedef float    v8f  __attribute__((ext_vector_type(8)));
typedef unsigned int v4u __attribute__((ext_vector_type(4)));
typedef int      v8i  __attribute__((ext_vector_type(8)));
typedef int      v4i  __attribute__((ext_vector_type(4)));

union V16H { v16h v; half_t e[16]; };
union V8F  { v8f  v; float  e[8];  };

#if defined(__HIP_DEVICE_COMPILE__) && __has_builtin(__builtin_amdgcn_tensor_load_to_lds) && __has_builtin(__builtin_amdgcn_s_wait_tensorcnt)
#define USE_TDM 1
#else
#define USE_TDM 0
#endif

// ---------------- workspace layout (float offsets) ----------------
// params region [0, 8192)
#define XD_O   ((size_t)0)        // 64   xd
#define HC_O   ((size_t)64)       // 768  h state
#define CS_O   ((size_t)832)      // 768  c state
#define ZB_O   ((size_t)1600)     // 2944 lstm gate pre-activations
#define OV_O   ((size_t)4544)     // 256  out_v
#define KRN_O  ((size_t)4800)     // 256  normalized read keys [r][d]
#define BRD_O  ((size_t)5056)     // 4    b_read
#define KWN_O  ((size_t)5060)     // 64   normalized write key
#define BWT_O  ((size_t)5124)     // 1    b_write
#define ERS_O  ((size_t)5125)     // 64   erase
#define WRV_O  ((size_t)5189)     // 64   write vector
#define FG_O   ((size_t)5253)     // 4    free gates
#define AG_O   ((size_t)5257)     // 1    alloc gate
#define WG_O   ((size_t)5258)     // 1    write gate
#define RM_O   ((size_t)5259)     // 12   read modes [m][r]
#define SV_O   ((size_t)5271)     // 4    s_r = sum p_j Wr[j,r]
#define TV_O   ((size_t)5275)     // 4    t_r = sum w_j Wr[j,r]
// big arrays
#define WAL_O  ((size_t)8192)     // 8192        W_alloc
#define WW_O   ((size_t)16384)    // 8192        w = new W_write
#define BF_O   ((size_t)24576)    // 8192*8      Bf = [Wr | w*Wr]
#define BB_O   ((size_t)90112)    // 8192*8      Bb = [Wr | (1-w)*Wr]
#define GP_O   ((size_t)155648)   // 16*8192*8   G partials (per col chunk)
#define HP_O   ((size_t)1204224)  // 32*8192*8   H partials (per row strip)
#define WF_O   ((size_t)3301376)  // 8192*4      W_fwd
#define WB_O   ((size_t)3334144)  // 8192*4      W_bwd
#define RP_O   ((size_t)3366912)  // 256*784     read-phase per-wave partials
// total ~3.57M floats (~14.3 MB) of d_ws

__device__ __forceinline__ float sigm(float x){ return 1.f/(1.f+__expf(-x)); }
__device__ __forceinline__ float wred(float x){
  #pragma unroll
  for (int m=16;m;m>>=1) x += __shfl_xor(x,m,32);
  return x;
}

// ---------------- 1) front: xd = x@dense_k+b, copy h,c ----------------
__global__ __launch_bounds__(256) void k_front(const float* __restrict__ x,
    const float* __restrict__ dk, const float* __restrict__ db,
    const float* __restrict__ h_in, const float* __restrict__ c_in, float* W)
{
  int t = threadIdx.x;
  if (t < WD) {
    float a = db[t];
    for (int k=0;k<IND;++k) a += x[k]*dk[(size_t)k*WD + t];
    W[XD_O + t] = a;
  }
  for (int i=t; i<CCn; i+=256) { W[HC_O+i] = h_in[i]; W[CS_O+i] = c_in[i]; }
}

// ---------------- 2) LSTM gemv: z = [xt;h] @ [lstm_k;lstm_r] + b ----------------
__global__ __launch_bounds__(256) void k_gemv(const float* __restrict__ xt,
    const float* __restrict__ hcur, const float* __restrict__ lk,
    const float* __restrict__ lr, const float* __restrict__ lb, float* __restrict__ z)
{
  __shared__ float xs[WD];
  __shared__ float hs[CCn];
  int t = threadIdx.x;
  if (t < WD) xs[t] = xt[t];
  for (int i=t; i<CCn; i+=256) hs[i] = hcur[i];
  __syncthreads();
  int c = blockIdx.x*256 + t;
  if (c < G4) {
    float a = lb[c];
    for (int k=0;k<WD;++k)  a += xs[k]*lk[(size_t)k*G4 + c];
    for (int k=0;k<CCn;++k) a += hs[k]*lr[(size_t)k*G4 + c];
    z[c] = a;
  }
}

__global__ __launch_bounds__(768) void k_gate(const float* __restrict__ z,
    float* __restrict__ h, float* __restrict__ c)
{
  int n = threadIdx.x;
  if (n < CCn) {
    float zi = z[n], zf = z[CCn+n], zg = z[2*CCn+n], zo = z[3*CCn+n];
    float cn = sigm(zf)*c[n] + sigm(zi)*tanhf(zg);
    float hn = sigm(zo)*tanhf(cn);
    c[n] = cn; h[n] = hn;
  }
}

// ---------------- 3) interface: out_v, iface parse ----------------
__global__ __launch_bounds__(1024) void k_iface(const float* __restrict__ Wo,
    const float* __restrict__ Wi, float* W)
{
  __shared__ float hs[CCn];
  __shared__ float ifr[471];
  __shared__ float nrm[5];
  int t = threadIdx.x;
  for (int i=t; i<CCn; i+=1024) hs[i] = W[HC_O+i];
  __syncthreads();
  if (t < OUTD) {
    float a = 0.f;
    for (int k=0;k<CCn;++k) a += hs[k]*Wo[(size_t)k*OUTD + t];
    W[OV_O + t] = a;
  } else if (t < CCn) {
    int c = t - OUTD;
    float a = 0.f;
    for (int k=0;k<CCn;++k) a += hs[k]*Wi[(size_t)k*471 + c];
    ifr[c] = a;
  }
  __syncthreads();
  if (t < 4) { float s=0; for(int d=0;d<64;++d){float v=ifr[t*64+d]; s+=v*v;} nrm[t]=s; }
  if (t == 4) { float s=0; for(int d=0;d<64;++d){float v=ifr[260+d]; s+=v*v;} nrm[4]=s; }
  __syncthreads();
  if (t < 256) W[KRN_O+t] = ifr[t]*rsqrtf(fmaxf(nrm[t>>6],1e-12f));
  else if (t < 320) { int d=t-256; W[KWN_O+d] = ifr[260+d]*rsqrtf(fmaxf(nrm[4],1e-12f)); }
  else if (t < 384) { int d=t-320; W[ERS_O+d] = sigm(ifr[325+d]); }
  else if (t < 448) { int d=t-384; W[WRV_O+d] = ifr[389+d]; }
  else if (t < 452) { int r=t-448; W[BRD_O+r] = 1.f + log1pf(__expf(ifr[256+r])); W[FG_O+r] = sigm(ifr[453+r]); }
  else if (t == 452) { W[BWT_O] = 1.f + log1pf(__expf(ifr[324])); W[AG_O] = sigm(ifr[457]); W[WG_O] = sigm(ifr[458]); }
  else if (t == 453) {
    for (int r=0;r<4;++r) {
      float e0=__expf(ifr[459+r]), e1=__expf(ifr[463+r]), e2=__expf(ifr[467+r]);
      float d = e0+e1+e2;
      W[RM_O+0*4+r]=e0/d; W[RM_O+1*4+r]=e1/d; W[RM_O+2*4+r]=e2/d;
    }
  }
}

// ---------------- 4) usage update + bitonic argsort + cumprod -> W_alloc ----------------
__global__ __launch_bounds__(1024) void k_usage_sort(const float* __restrict__ usage,
    const float* __restrict__ ww_in, const float* __restrict__ wr_in, float* W)
{
  __shared__ float key[NN];
  __shared__ unsigned short idx[NN];
  int tid = threadIdx.x;
  float fg0=W[FG_O+0], fg1=W[FG_O+1], fg2=W[FG_O+2], fg3=W[FG_O+3];
  for (int i=tid;i<NN;i+=1024) {
    float u = usage[i], wv = ww_in[i];
    float ret = (1.f-fg0*wr_in[(size_t)i*4+0])*(1.f-fg1*wr_in[(size_t)i*4+1])
              * (1.f-fg2*wr_in[(size_t)i*4+2])*(1.f-fg3*wr_in[(size_t)i*4+3]);
    key[i] = (u + wv - u*wv)*ret;
    idx[i] = (unsigned short)i;
  }
  __syncthreads();
  for (int k=2;k<=NN;k<<=1) {
    for (int j=k>>1;j>0;j>>=1) {
      for (int t=tid;t<NN;t+=1024) {
        int ixj = t ^ j;
        if (ixj > t) {
          bool up = ((t & k) == 0);
          float a=key[t], b=key[ixj];
          unsigned short ia=idx[t], ib=idx[ixj];
          bool sw = up ? (a>b || (a==b && ia>ib)) : (a<b || (a==b && ia<ib));
          if (sw) { key[t]=b; key[ixj]=a; idx[t]=ib; idx[ixj]=ia; }
        }
      }
      __syncthreads();
    }
  }
  float myu[8];
  #pragma unroll
  for (int q=0;q<8;++q) myu[q] = key[tid + 1024*q];
  for (int off=1; off<NN; off<<=1) {
    float tmp[8];
    #pragma unroll
    for (int q=0;q<8;++q) { int i=tid+1024*q; tmp[q] = (i>=off)? key[i-off] : 1.f; }
    __syncthreads();
    #pragma unroll
    for (int q=0;q<8;++q) key[tid+1024*q] *= tmp[q];
    __syncthreads();
  }
  #pragma unroll
  for (int q=0;q<8;++q) {
    int i = tid + 1024*q;
    float excl = (i>0)? key[i-1] : 1.f;
    W[WAL_O + idx[i]] = (1.f - myu[q])*excl;
  }
}

// ---------------- 5) write weights + s/t vectors ----------------
__global__ __launch_bounds__(1024) void k_wwrite(const float* __restrict__ M,
    const float* __restrict__ Wprec, const float* __restrict__ Wread, float* W)
{
  __shared__ float kw[64];
  __shared__ float red[1024];
  __shared__ float Ssum;
  int tid = threadIdx.x;
  if (tid < 64) kw[tid] = W[KWN_O+tid];
  __syncthreads();
  float bw = W[BWT_O], ag = W[AG_O], wg = W[WG_O];
  float eq[8];
  #pragma unroll
  for (int q=0;q<8;++q) {
    int i = tid + 1024*q;
    const float4* mp = (const float4*)(M + (size_t)i*64);
    float dot=0.f, ss=0.f;
    for (int b=0;b<16;++b) {
      float4 v = mp[b];
      int d = b*4;
      dot += v.x*kw[d]+v.y*kw[d+1]+v.z*kw[d+2]+v.w*kw[d+3];
      ss  += v.x*v.x+v.y*v.y+v.z*v.z+v.w*v.w;
    }
    float cosb = dot*rsqrtf(fmaxf(ss,1e-12f))*bw;
    eq[q] = __expf(cosb);
  }
  float ps = eq[0]+eq[1]+eq[2]+eq[3]+eq[4]+eq[5]+eq[6]+eq[7];
  red[tid]=ps; __syncthreads();
  for (int s=512;s>0;s>>=1){ if(tid<s) red[tid]+=red[tid+s]; __syncthreads(); }
  if (tid==0) Ssum = red[0];
  __syncthreads();
  float sacc[4]={0,0,0,0}, tacc[4]={0,0,0,0};
  #pragma unroll
  for (int q=0;q<8;++q) {
    int i = tid + 1024*q;
    float wi = wg*(ag*W[WAL_O+i] + (1.f-ag)*eq[q]/Ssum);
    W[WW_O+i] = wi;
    float pi = Wprec[i];
    #pragma unroll
    for (int r=0;r<4;++r) { float wr = Wread[(size_t)i*4+r]; sacc[r]+=pi*wr; tacc[r]+=wi*wr; }
  }
  for (int r=0;r<8;++r) {
    float v = (r<4)? sacc[r] : tacc[r-4];
    __syncthreads(); red[tid]=v; __syncthreads();
    for (int s=512;s>0;s>>=1){ if(tid<s) red[tid]+=red[tid+s]; __syncthreads(); }
    if (tid==0) { if (r<4) W[SV_O+r]=red[0]; else W[TV_O+r-4]=red[0]; }
  }
}

// ---------------- 6) build Bf/Bb (8 columns each) ----------------
__global__ __launch_bounds__(256) void k_prepB(const float* __restrict__ Wread, float* W)
{
  int i = blockIdx.x*256 + threadIdx.x;
  float wi = W[WW_O+i];
  #pragma unroll
  for (int r=0;r<4;++r) {
    float wr = Wread[(size_t)i*4+r];
    W[BF_O + (size_t)i*8 + r]     = wr;
    W[BF_O + (size_t)i*8 + 4 + r] = wi*wr;
    W[BB_O + (size_t)i*8 + r]     = wr;
    W[BB_O + (size_t)i*8 + 4 + r] = (1.f-wi)*wr;
  }
}

// ---------------- 7) single pass over L: G = L@Bf, H = L^T@Bb (WMMA + TDM) --------
// grid: (16 col chunks of 512) x (32 row strips of 256). block = 256 threads = 8 waves.
// LDS tile: 64x64 f32 with 65-float row stride (TDM pad: interval 64 DW, amount 1 DW),
// double-buffered; wave 0 drives the Tensor Data Mover one panel ahead.
// B slices stored transposed [n][K] with rows n=8..15 zeroed -> unconditional,
// fragment loads of two contiguous 8-half runs (no divergent selects).
#define LTSTR 65
#define BSTR  72

#if USE_TDM
__device__ __forceinline__ void tdm_issue(const float* gsrc, float* ldsdst)
{
  unsigned long long ga = (unsigned long long)(const void*)gsrc;
  unsigned lo = (unsigned)(unsigned long long)(void*)ldsdst;   // low 32 bits = LDS offset
  v4u g0 = { 1u,                                   // count=1, user descriptor
             lo,                                   // lds_addr
             (unsigned)(ga & 0xFFFFFFFFu),         // global_addr[31:0]
             (unsigned)((ga >> 32) & 0x01FFFFFFu) | (2u << 30) }; // addr[56:32] | type=2
  v8i g1 = { (int)((2u<<16) | (1u<<20) | (5u<<22)), // data_size=4B, pad_en, interval=64DW (amount=1DW)
             (int)(0x2000u << 16),                  // tensor_dim0 = 8192 (lo16)
             (int)(0x2000u << 16),                  // tensor_dim0 hi=0 | tensor_dim1 lo16 = 8192
             (int)(64u << 16),                      // tensor_dim1 hi=0 | tile_dim0 = 64
             64,                                    // tile_dim1 = 64, tile_dim2 = 0
             8192, 0, 0 };                          // tensor_dim0_stride = 8192
  v4i gz = {0,0,0,0};
  v8i gz8 = {0,0,0,0,0,0,0,0};
  __builtin_amdgcn_tensor_load_to_lds(g0, g1, gz, gz, gz8, 0);
}
#endif

__global__ __launch_bounds__(256) void k_lpass(const float* __restrict__ L,
    float* __restrict__ W)
{
  __shared__ float  LtA[2][64*LTSTR];   // 33.3 KB
  __shared__ float  Hacc[512][8];       // 16 KB
  __shared__ half_t BfT[16][BSTR];      // [n][K], rows 8..15 stay zero
  __shared__ half_t BbT[16][BSTR];

  const float* Bf = W + BF_O;
  const float* Bb = W + BB_O;
  float* Gpart = W + GP_O;
  float* Hpart = W + HP_O;

  const int chunk = blockIdx.x;   // col chunk (512 cols)
  const int strip = blockIdx.y;   // row strip (256 rows)
  const int tid = threadIdx.x;
  const int wave = tid >> 5;
  const int lane = tid & 31;
  const bool lolane = (lane < 16);
  const int lm = lane & 15;

  for (int t=tid; t<512*8; t+=256) ((float*)Hacc)[t] = 0.f;
  for (int t=tid; t<16*BSTR; t+=256) { ((half_t*)BfT)[t] = (half_t)0; ((half_t*)BbT)[t] = (half_t)0; }

#if USE_TDM
  if (wave == 0)   // prologue: panel 0 -> buf 0
    tdm_issue(L + (size_t)(strip*256)*NN + chunk*512, LtA[0]);
#endif

  V8F accF;
  for (int p=0; p<32; ++p) {
    const int rp = p >> 3, cp = p & 7;
    const int buf = p & 1;
    const int i0 = strip*256 + rp*64;
    const int j0 = chunk*512 + cp*64;
#if USE_TDM
    if (wave == 0) __builtin_amdgcn_s_wait_tensorcnt(0);
    __syncthreads();            // tile[buf] ready; previous compute done
    if (wave == 0 && p < 31) {
      const int pn = p + 1, rpn = pn >> 3, cpn = pn & 7;
      tdm_issue(L + (size_t)(strip*256 + rpn*64)*NN + chunk*512 + cpn*64, LtA[buf^1]);
    }
#else
    __syncthreads();            // previous compute done; safe to restage
    {   // manual staging: 4 independent b128 loads per thread, then scalar LDS stores
      int r0 = tid >> 2;
      int c0 = (tid & 3) << 4;
      const float4* src = (const float4*)(L + (size_t)(i0 + r0)*NN + j0 + c0);
      float4 v0 = src[0], v1 = src[1], v2 = src[2], v3 = src[3];
      float* dst = LtA[buf] + r0*LTSTR + c0;
      dst[0]=v0.x; dst[1]=v0.y; dst[2]=v0.z; dst[3]=v0.w;
      dst[4]=v1.x; dst[5]=v1.y; dst[6]=v1.z; dst[7]=v1.w;
      dst[8]=v2.x; dst[9]=v2.y; dst[10]=v2.z; dst[11]=v2.w;
      dst[12]=v3.x; dst[13]=v3.y; dst[14]=v3.z; dst[15]=v3.w;
    }
#endif
    // stage B slices (f16, transposed [n][K]; rows 8..15 remain zero)
    for (int t=tid; t<512; t+=256)
      BfT[t & 7][t >> 3] = (half_t)Bf[(size_t)(j0 + (t>>3))*8 + (t & 7)];
    if (cp == 0)
      for (int t=tid; t<512; t+=256)
        BbT[t & 7][t >> 3] = (half_t)Bb[(size_t)(i0 + (t>>3))*8 + (t & 7)];
    if (cp == 0 && wave < 4) {
      #pragma unroll
      for (int q=0;q<8;++q) accF.e[q]=0.f;
    }
    __syncthreads();            // tile + B slices visible
    const float* Lt = LtA[buf];
    if (wave < 4) {             // FWD: rows 16*wave..+15, K = panel cols
      V16H a, b;
      #pragma unroll
      for (int kk=0; kk<2; ++kk) {
        const half_t* bp = &BfT[lm][kk*32 + (lolane?0:8)];
        #pragma unroll
        for (int h=0; h<16; ++h) {
          int k = ((h & 8)?16:0) + (lolane?0:8) + (((h>>1)&3)<<1) + (h&1);
          a.e[h] = (half_t)Lt[(wave*16 + lm)*LTSTR + kk*32 + k];
          b.e[h] = bp[h + ((h & 8)?8:0)];
        }
        accF.v = __builtin_amdgcn_wmma_f32_16x16x32_f16(false, a.v, false, b.v,
                                                        (short)0, accF.v, false, false);
      }
      if (cp == 7 && lm < 8) {  // end of this rp: write G partial rows
        #pragma unroll
        for (int v=0; v<8; ++v) {
          int m = v + (lolane?0:8);
          int row = i0 + wave*16 + m;
          Gpart[((size_t)chunk*NN + row)*8 + lm] = accF.e[v];
        }
      }
    } else {                    // BWD: j group (wave-4)*16..+15, K = panel rows
      V8F cb;
      #pragma unroll
      for (int q=0;q<8;++q) cb.e[q]=0.f;
      V16H a, b;
      const int jg = (wave-4)*16;
      #pragma unroll
      for (int kk=0; kk<2; ++kk) {
        const half_t* bp = &BbT[lm][kk*32 + (lolane?0:8)];
        #pragma unroll
        for (int h=0; h<16; ++h) {
          int k = ((h & 8)?16:0) + (lolane?0:8) + (((h>>1)&3)<<1) + (h&1);
          a.e[h] = (half_t)Lt[(kk*32 + k)*LTSTR + jg + lm];
          b.e[h] = bp[h + ((h & 8)?8:0)];
        }
        cb.v = __builtin_amdgcn_wmma_f32_16x16x32_f16(false, a.v, false, b.v,
                                                      (short)0, cb.v, false, false);
      }
      if (lm < 8) {
        #pragma unroll
        for (int v=0; v<8; ++v) {
          int m = v + (lolane?0:8);
          Hacc[cp*64 + jg + m][lm] += cb.e[v];
        }
      }
    }
  }
  __syncthreads();
  for (int t=tid; t<512*8; t+=256) {
    int j = t >> 3, n = t & 7;
    Hpart[((size_t)strip*NN + chunk*512 + j)*8 + n] = Hacc[j][n];
  }
}

// ---------------- 8) reduce partials -> W_fwd, W_bwd ----------------
__global__ __launch_bounds__(256) void k_combine(const float* __restrict__ L,
    const float* __restrict__ Wprec, const float* __restrict__ Wread, float* W)
{
  int i = blockIdx.x*256 + threadIdx.x;
  float g8[8] = {0,0,0,0,0,0,0,0};
  for (int c=0;c<16;++c) {
    const float4* gp = (const float4*)(W + GP_O + ((size_t)c*NN + i)*8);
    float4 a = gp[0], b = gp[1];
    g8[0]+=a.x; g8[1]+=a.y; g8[2]+=a.z; g8[3]+=a.w;
    g8[4]+=b.x; g8[5]+=b.y; g8[6]+=b.z; g8[7]+=b.w;
  }
  float h8[8] = {0,0,0,0,0,0,0,0};
  for (int s=0;s<32;++s) {
    const float4* hp = (const float4*)(W + HP_O + ((size_t)s*NN + i)*8);
    float4 a = hp[0], b = hp[1];
    h8[0]+=a.x; h8[1]+=a.y; h8[2]+=a.z; h8[3]+=a.w;
    h8[4]+=b.x; h8[5]+=b.y; h8[6]+=b.z; h8[7]+=b.w;
  }
  float wi = W[WW_O+i], pi = Wprec[i];
  float Lii = L[(size_t)i*NN + i];
  float di = (1.f - 2.f*wi)*Lii + wi*pi;
  #pragma unroll
  for (int r=0;r<4;++r) {
    float wr = Wread[(size_t)i*4+r];
    W[WF_O + (size_t)i*4 + r] = (1.f-wi)*g8[r] - g8[4+r] + wi*W[SV_O+r] - di*wr;
    W[WB_O + (size_t)i*4 + r] = h8[4+r] - wi*h8[r] + pi*W[TV_O+r] - di*wr;
  }
}

// ---------------- 9) read phase: per-wave partials of Se, M_new·{Wb,e,Wf} -------
__global__ __launch_bounds__(256) void k_read1(const float* __restrict__ M, float* W)
{
  int wave = threadIdx.x >> 5, lane = threadIdx.x & 31;
  int g = blockIdx.x*8 + wave;
  float e0 = W[ERS_O+lane], e1 = W[ERS_O+32+lane];
  float wv0 = W[WRV_O+lane], wv1 = W[WRV_O+32+lane];
  float kr0[4], kr1[4], brd[4];
  #pragma unroll
  for (int r=0;r<4;++r) {
    kr0[r]=W[KRN_O + r*64 + lane]; kr1[r]=W[KRN_O + r*64 + 32 + lane];
    brd[r]=W[BRD_O+r];
  }
  float a0[2][4]={{0}}, a1[2][4]={{0}}, a2[2][4]={{0}}, se[4]={0,0,0,0};
  for (int it=0; it<32; ++it) {
    int i = g*32 + it;
    float wi = W[WW_O+i];
    float m0 = M[(size_t)i*64 + lane], m1 = M[(size_t)i*64 + 32 + lane];
    float mn0 = m0*(1.f - wi*e0) + wi*wv0;
    float mn1 = m1*(1.f - wi*e1) + wi*wv1;
    float inv = rsqrtf(fmaxf(wred(mn0*mn0 + mn1*mn1), 1e-12f));
    #pragma unroll
    for (int r=0;r<4;++r) {
      float dot = wred(mn0*kr0[r] + mn1*kr1[r]);
      float er = __expf(dot*inv*brd[r]);
      float wf = W[WF_O + (size_t)i*4 + r];
      float wb = W[WB_O + (size_t)i*4 + r];
      a0[0][r]+=mn0*wb; a0[1][r]+=mn1*wb;
      a1[0][r]+=mn0*er; a1[1][r]+=mn1*er;
      a2[0][r]+=mn0*wf; a2[1][r]+=mn1*wf;
      se[r]+=er;
    }
  }
  float* base = W + RP_O + (size_t)g*784;
  #pragma unroll
  for (int r=0;r<4;++r) {
    base[       lane*4 + r] = a0[0][r]; base[       (lane+32)*4 + r] = a0[1][r];
    base[256 +  lane*4 + r] = a1[0][r]; base[256 +  (lane+32)*4 + r] = a1[1][r];
    base[512 +  lane*4 + r] = a2[0][r]; base[512 +  (lane+32)*4 + r] = a2[1][r];
  }
  if (lane == 0) {
    #pragma unroll
    for (int r=0;r<4;++r) base[768 + r] = se[r];
  }
}

// ---------------- 10) final: read_v, y = out_v + read_v @ W_read_out ----------
__global__ __launch_bounds__(256) void k_back(const float* __restrict__ Wro,
    float* W, float* __restrict__ y)
{
  __shared__ float A0s[256], A1s[256], A2s[256], Ses[4], rv[256];
  int t = threadIdx.x;
  float s0=0.f, s1=0.f, s2=0.f;
  for (int g=0; g<256; ++g) {
    const float* base = W + RP_O + (size_t)g*784;
    s0 += base[t]; s1 += base[256+t]; s2 += base[512+t];
  }
  A0s[t]=s0; A1s[t]=s1; A2s[t]=s2;
  if (t < 4) { float s=0.f; for (int g=0; g<256; ++g) s += W[RP_O + (size_t)g*784 + 768 + t]; Ses[t]=s; }
  __syncthreads();
  {
    int r = t >> 6, d = t & 63;
    int e = d*4 + r;
    rv[t] = W[RM_O+0*4+r]*A0s[e] + W[RM_O+1*4+r]*A1s[e]/Ses[r] + W[RM_O+2*4+r]*A2s[e];
  }
  __syncthreads();
  float a = W[OV_O + t];
  for (int k=0;k<256;++k) a += rv[k]*Wro[(size_t)k*256 + t];
  y[t] = a;
}

// ---------------- launch ----------------
extern "C" void kernel_launch(void* const* d_in, const int* in_sizes, int n_in,
                              void* d_out, int out_size, void* d_ws, size_t ws_size,
                              hipStream_t stream) {
  (void)in_sizes; (void)n_in; (void)out_size; (void)ws_size;
  const float* x       = (const float*)d_in[0];
  const float* dense_k = (const float*)d_in[1];
  const float* dense_b = (const float*)d_in[2];
  const float* lstm_k  = (const float*)d_in[3];
  const float* lstm_r  = (const float*)d_in[4];
  const float* lstm_b  = (const float*)d_in[5];
  const float* W_out   = (const float*)d_in[6];
  const float* W_if    = (const float*)d_in[7];
  const float* W_ro    = (const float*)d_in[8];
  const float* h_in    = (const float*)d_in[9];
  const float* c_in    = (const float*)d_in[10];
  const float* M       = (const float*)d_in[11];
  const float* usage   = (const float*)d_in[12];
  const float* L       = (const float*)d_in[13];
  const float* W_prec  = (const float*)d_in[14];
  const float* W_read  = (const float*)d_in[15];
  const float* W_write = (const float*)d_in[16];
  const float* read_v  = (const float*)d_in[17];
  float* W = (float*)d_ws;
  float* y = (float*)d_out;

  k_front<<<1,256,0,stream>>>(x, dense_k, dense_b, h_in, c_in, W);
  for (int t=0;t<5;++t) {
    const float* xt = (t==0) ? (W + XD_O) : (read_v + (size_t)(t-1)*WD);
    k_gemv<<<12,256,0,stream>>>(xt, W+HC_O, lstm_k, lstm_r, lstm_b, W+ZB_O);
    k_gate<<<1,768,0,stream>>>(W+ZB_O, W+HC_O, W+CS_O);
  }
  k_iface<<<1,1024,0,stream>>>(W_out, W_if, W);
  k_usage_sort<<<1,1024,0,stream>>>(usage, W_write, W_read, W);
  k_wwrite<<<1,1024,0,stream>>>(M, W_prec, W_read, W);
  k_prepB<<<32,256,0,stream>>>(W_read, W);
  k_lpass<<<dim3(16,32),256,0,stream>>>(L, W);
  k_combine<<<32,256,0,stream>>>(L, W_prec, W_read, W);
  k_read1<<<32,256,0,stream>>>(M, W);
  k_back<<<1,256,0,stream>>>(W_ro, W, y);
}
